// CLA_19636590478164
// MI455X (gfx1250) — compile-verified, matrix operand order checked by reading, not run on hardware
//
#include <hip/hip_runtime.h>

typedef __attribute__((ext_vector_type(16))) _Float16 v16h;
typedef __attribute__((ext_vector_type(8)))  float    v8f;

#define Hdim 64
#define Wdim 64
#define Cdim 256
#define BATCH 4
#define KTOT 16
#define HW 4096
#define NIMG 16              // ref_num * batch
#define NPIX (NIMG * HW)     // 65536

// ---------------- Kernel 1: kf[pix][d] = sum_c w_ref[d][c]*key[img][c][hw] + b_ref[d]
#define TM 64
#define TN 128
#define LSTR (Cdim + 8)      // padded LDS row stride (f16 units)

__device__ __forceinline__ int iclamp(int v, int lo, int hi) {
    return v < lo ? lo : (v > hi ? hi : v);
}

__device__ __forceinline__ v8f wmma_f16(const v16h& a, const v16h& b, const v8f& c) {
    return __builtin_amdgcn_wmma_f32_16x16x32_f16(false, a, false, b,
                                                  (short)0, c, false, false);
}

__global__ __launch_bounds__(256) void kf_gemm_kernel(const float* __restrict__ key,
                                                      const float* __restrict__ w_ref,
                                                      const float* __restrict__ b_ref,
                                                      float* __restrict__ kf) {
    __shared__ _Float16 As[TM * LSTR];   // [pixel][c]   33.8 KB
    __shared__ _Float16 Bs[TN * LSTR];   // [d][c]       67.6 KB

    const int tid = threadIdx.x;
    const int p0  = blockIdx.x * TM;         // pixel tile start (never straddles an image)
    const int rb  = p0 >> 12;                // image index (r*4+b)
    const int hw0 = p0 & (HW - 1);
    const int d0  = blockIdx.y * TN;

    const float* keyBase = key + (size_t)rb * Cdim * HW + hw0;

    // Stage A: 64 pixels x 256 channels. float4 along contiguous pixel dim.
    {
        const int p4 = (tid & 15) * 4;       // 0..60
        const int c0 = tid >> 4;             // 0..15
        #pragma unroll
        for (int i = 0; i < 16; ++i) {
            int c = i * 16 + c0;
            const float4 v = *(const float4*)(keyBase + (size_t)c * HW + p4);
            As[(p4 + 0) * LSTR + c] = (_Float16)v.x;
            As[(p4 + 1) * LSTR + c] = (_Float16)v.y;
            As[(p4 + 2) * LSTR + c] = (_Float16)v.z;
            As[(p4 + 3) * LSTR + c] = (_Float16)v.w;
        }
    }
    // Stage B: 128 d-rows x 256 channels of w_ref. float4 along c, packed f16 b64 stores.
    {
        const int c4 = (tid & 63) * 4;       // 0..252
        const int r0 = tid >> 6;             // 0..3
        #pragma unroll
        for (int i = 0; i < 32; ++i) {
            int r = i * 4 + r0;
            const float4 v = *(const float4*)(w_ref + (size_t)(d0 + r) * Cdim + c4);
            union { _Float16 h[4]; unsigned long long u; } pk;
            pk.h[0] = (_Float16)v.x; pk.h[1] = (_Float16)v.y;
            pk.h[2] = (_Float16)v.z; pk.h[3] = (_Float16)v.w;
            *(unsigned long long*)(Bs + r * LSTR + c4) = pk.u;   // 8B-aligned (LSTR*2 % 8 == 0)
        }
    }
    __syncthreads();

    const int lane   = tid & 31;
    const int wv     = tid >> 5;          // 8 waves: 2 (M) x 4 (N), each 32x32 output
    const int m_base = (wv & 1) * 32;
    const int n_base = (wv >> 1) * 32;
    const int mn     = lane & 15;         // A: row M, B: col N, D: col N
    const int half   = lane >> 4;         // K half for A/B operands, M half for D

    const unsigned int* A0 = (const unsigned int*)(As + (m_base +      mn) * LSTR);
    const unsigned int* A1 = (const unsigned int*)(As + (m_base + 16 + mn) * LSTR);
    const unsigned int* B0 = (const unsigned int*)(Bs + (n_base +      mn) * LSTR);
    const unsigned int* B1 = (const unsigned int*)(Bs + (n_base + 16 + mn) * LSTR);

    v8f acc00 = {}, acc01 = {}, acc10 = {}, acc11 = {};
    #pragma unroll
    for (int kk = 0; kk < 8; ++kk) {       // K = 256 in steps of 32
        const int kbase = kk * 16;         // u32 units
        union { unsigned int u[8]; v16h h; } a0, a1, b0, b1;
        // 16-bit operand layout: u32 element j holds K pair 2j + 8*half (+8 for j>=4)
        #pragma unroll
        for (int j = 0; j < 8; ++j) {
            int off = kbase + j + 4 * half + ((j >= 4) ? 4 : 0);
            a0.u[j] = A0[off];
            a1.u[j] = A1[off];
            b0.u[j] = B0[off];
            b1.u[j] = B1[off];
        }
        acc00 = wmma_f16(a0.h, b0.h, acc00);
        acc01 = wmma_f16(a0.h, b1.h, acc01);
        acc10 = wmma_f16(a1.h, b0.h, acc10);
        acc11 = wmma_f16(a1.h, b1.h, acc11);
    }

    const int col0 = d0 + n_base + mn;
    const int col1 = col0 + 16;
    const float bv0 = b_ref[col0];
    const float bv1 = b_ref[col1];
    #pragma unroll
    for (int v = 0; v < 8; ++v) {          // D: VGPR v -> M = v + 8*half
        int prow0 = p0 + m_base + v + 8 * half;
        int prow1 = prow0 + 16;
        kf[(size_t)prow0 * Cdim + col0] = acc00[v] + bv0;
        kf[(size_t)prow0 * Cdim + col1] = acc01[v] + bv1;
        kf[(size_t)prow1 * Cdim + col0] = acc10[v] + bv0;
        kf[(size_t)prow1 * Cdim + col1] = acc11[v] + bv1;
    }
}

// ---------------- Kernel 2: attn logits + softmax, and offsets, per pixel
__global__ __launch_bounds__(256) void proj_kernel(const float* __restrict__ query,
                                                   const float* __restrict__ w_off,
                                                   const float* __restrict__ b_off,
                                                   const float* __restrict__ w_attn,
                                                   const float* __restrict__ b_attn,
                                                   float* __restrict__ attn_ws,
                                                   float* __restrict__ off_ws) {
    __shared__ float WA[KTOT * Cdim];        // 16 KB
    __shared__ float WO[2 * KTOT * Cdim];    // 32 KB
    const int tid = threadIdx.x;
    for (int i = tid; i < KTOT * Cdim; i += 256)      WA[i] = w_attn[i];
    for (int i = tid; i < 2 * KTOT * Cdim; i += 256)  WO[i] = w_off[i];
    __syncthreads();

    const int pix  = blockIdx.x * 256 + tid;
    const int bimg = pix >> 12;
    const int hw   = pix & (HW - 1);
    const float* q = query + (size_t)bimg * Cdim * HW + hw;

    float accA[16];
    float accO[32];
    #pragma unroll
    for (int o = 0; o < 16; ++o) accA[o] = 0.f;
    #pragma unroll
    for (int o = 0; o < 32; ++o) accO[o] = 0.f;

    for (int c = 0; c < Cdim; ++c) {
        float qv = q[(size_t)c * HW];
        #pragma unroll
        for (int o = 0; o < 16; ++o) accA[o] += qv * WA[o * Cdim + c];
        #pragma unroll
        for (int o = 0; o < 32; ++o) accO[o] += qv * WO[o * Cdim + c];
    }

    // softmax over K=16
    float mx = -3.0e38f;
    #pragma unroll
    for (int k = 0; k < 16; ++k) {
        accA[k] += b_attn[k];
        mx = fmaxf(mx, accA[k]);
    }
    float sum = 0.f;
    #pragma unroll
    for (int k = 0; k < 16; ++k) {
        accA[k] = __expf(accA[k] - mx);
        sum += accA[k];
    }
    float inv = 1.f / sum;
    #pragma unroll
    for (int k = 0; k < 16; ++k)
        attn_ws[((size_t)bimg * 16 + k) * HW + hw] = accA[k] * inv;
    #pragma unroll
    for (int o = 0; o < 32; ++o)
        off_ws[((size_t)bimg * 32 + o) * HW + hw] = accO[o] + b_off[o];
}

// ---------------- Kernel 3: deformable bilinear gather + attention-weighted sum
__global__ __launch_bounds__(256) void sample_kernel(const float* __restrict__ kf,
                                                     const float* __restrict__ attn_ws,
                                                     const float* __restrict__ off_ws,
                                                     float* __restrict__ out) {
    __shared__ float s_w[KTOT][4];
    __shared__ int   s_idx[KTOT][4];

    const int pix  = blockIdx.x;
    const int bimg = pix >> 12;
    const int hw   = pix & (HW - 1);
    const int hy   = hw >> 6;
    const int wx   = hw & 63;
    const int tid  = threadIdx.x;

    if (tid < KTOT) {
        const int k  = tid;
        float dy = off_ws[((size_t)bimg * 32 + k) * HW + hw];
        float dx = off_ws[((size_t)bimg * 32 + k + 1) * HW + hw];
        float av = attn_ws[((size_t)bimg * 16 + k) * HW + hw];
        float py = (float)hy + dy;
        float px = (float)wx + dx;
        float y0f = floorf(py), x0f = floorf(px);
        float ly = py - y0f,   lx = px - x0f;
        int y0 = (int)y0f, x0 = (int)x0f;
        int base = ((k >> 2) * BATCH + bimg) * HW;   // ridx = k // key_num
        float wy[2]  = {1.f - ly, ly};
        float wxv[2] = {1.f - lx, lx};
        #pragma unroll
        for (int cy = 0; cy < 2; ++cy) {
            #pragma unroll
            for (int cx = 0; cx < 2; ++cx) {
                int yy = y0 + cy, xx = x0 + cx;
                bool valid = (yy >= 0) && (yy < Hdim) && (xx >= 0) && (xx < Wdim);
                int yc = iclamp(yy, 0, Hdim - 1);
                int xc = iclamp(xx, 0, Wdim - 1);
                s_idx[k][cy * 2 + cx] = base + yc * Wdim + xc;
                s_w[k][cy * 2 + cx]   = valid ? (av * wy[cy] * wxv[cx]) : 0.f;
            }
        }
    }
    __syncthreads();

    float acc = 0.f;
    #pragma unroll
    for (int k = 0; k < KTOT; ++k) {
        #pragma unroll
        for (int cn = 0; cn < 4; ++cn) {
            acc += s_w[k][cn] * kf[(size_t)s_idx[k][cn] * Cdim + tid];
        }
    }
    out[((size_t)bimg * Cdim + tid) * HW + hw] = acc;
}

extern "C" void kernel_launch(void* const* d_in, const int* in_sizes, int n_in,
                              void* d_out, int out_size, void* d_ws, size_t ws_size,
                              hipStream_t stream) {
    const float* query  = (const float*)d_in[0];
    const float* key    = (const float*)d_in[1];
    const float* w_off  = (const float*)d_in[2];
    const float* b_off  = (const float*)d_in[3];
    const float* w_attn = (const float*)d_in[4];
    const float* b_attn = (const float*)d_in[5];
    const float* w_ref  = (const float*)d_in[6];
    const float* b_ref  = (const float*)d_in[7];
    float* out = (float*)d_out;

    // workspace layout (floats): kf (64 MiB) | attn (1 MiB) | off (2 MiB)
    float* kf      = (float*)d_ws;
    float* attn_ws = kf + (size_t)NPIX * Cdim;
    float* off_ws  = attn_ws + (size_t)BATCH * KTOT * HW;

    dim3 g1(NPIX / TM, Cdim / TN);   // 1024 x 2 blocks, 256 threads (8 waves)
    kf_gemm_kernel<<<g1, 256, 0, stream>>>(key, w_ref, b_ref, kf);

    proj_kernel<<<(BATCH * HW) / 256, 256, 0, stream>>>(query, w_off, b_off,
                                                        w_attn, b_attn,
                                                        attn_ws, off_ws);

    sample_kernel<<<BATCH * HW, 256, 0, stream>>>(kf, attn_ws, off_ws, out);
}
